// RelationPredictor_56470230007955
// MI455X (gfx1250) — matrix operand order repaired and checked by reading, not compiled, not run on hardware
//
#include <hip/hip_runtime.h>
#include <hip/hip_bf16.h>

// Problem constants (match reference)
#define BB   8192
#define SS   64
#define DD   1024
#define K1   3072            // concatenated [head;prep;child] reduction dim
#define PJ   512             // PROJ
#define OUTN 32
#define KC1  (K1 / 32)       // 96  K-chunks of GEMM1
#define KC2  (PJ / 32)       // 16  K-chunks of GEMM2/3
#define NT1  (PJ / 16)       // 32  N-tiles of GEMM1/2
#define NTS  (OUTN / 16)     // 2   N-tiles of GEMM3
#define MT   32              // batch rows per block (2 WMMA M-halves)

// Packed-weight sizes (in u32; each u32 = 2 bf16)
#define WCAT_U ((size_t)NT1 * KC1 * 32 * 8)  // 786432
#define WHID_U ((size_t)NT1 * KC2 * 32 * 8)  // 131072
#define WSC_U  ((size_t)NTS * KC2 * 32 * 8)  // 8192

typedef __attribute__((ext_vector_type(16))) __bf16 v16bf;
typedef __attribute__((ext_vector_type(8)))  float  v8f;

union Frag {
    unsigned u[8];
    uint4    q[2];
    v16bf    v;
};

// f32 -> bf16 round-to-nearest, packed pair (a -> [15:0], b -> [31:16]).
// One add per value + a single v_perm_b32 byte-select of both high halves.
__device__ __forceinline__ unsigned bf16pk(float a, float b) {
    unsigned ua = __float_as_uint(a) + 0x8000u;
    unsigned ub = __float_as_uint(b) + 0x8000u;
    return __builtin_amdgcn_perm(ub, ua, 0x07060302u);  // [ua.hi16 | ub.hi16]
}

// branch-free tanh: sign(x) * (1 - e) / (1 + e), e = exp(-2|x|)
__device__ __forceinline__ float fast_tanh(float x) {
    float ax = __builtin_fabsf(x);
    float e  = __expf(-2.f * ax);
    float r  = (1.f - e) * __builtin_amdgcn_rcpf(1.f + e);
    return __builtin_copysignf(r, x);
}

// A-fragment swizzle: pair p (0..15) of local row ml (0..15) lives at
// lane = ml + 16*(p>>3), dword j = p&7.
// Slot (in u32) within a [lane][8] fragment block:
__device__ __forceinline__ int aslot(int ml, int p) {
    return (ml + ((p >> 3) << 4)) * 8 + (p & 7);
}

// ---------------------------------------------------------------------------
// Weight packing: emit B-matrix WMMA fragments, fragment-major.
// Fragment (nt, kc): lane l holds N = nt*16 + (l&15), K-half selected by l>=16,
// u32 j holds bf16 pair (K = kbase + 2j, kbase + 2j + 1).
// dst index == ((nt*KC + kc)*32 + lane)*8 + j  == flat thread id.
// ---------------------------------------------------------------------------
__global__ __launch_bounds__(256) void pack_cat_k(const float* __restrict__ Wh,
                                                  const float* __restrict__ Wp,
                                                  const float* __restrict__ Wc,
                                                  unsigned* __restrict__ dst) {
    int g    = blockIdx.x * 256 + threadIdx.x;
    int j    = g & 7;
    int lane = (g >> 3) & 31;
    int frag = g >> 8;
    int kc   = frag % KC1;
    int nt   = frag / KC1;
    int n    = nt * 16 + (lane & 15);
    int k0   = kc * 32 + ((lane >> 4) << 4) + 2 * j;   // even; never crosses 1024-boundary
    const float* W0 = (k0 < 1024) ? Wh : ((k0 < 2048) ? Wp : Wc);
    int kk = k0 & 1023;
    dst[g] = bf16pk(W0[kk * PJ + n], W0[(kk + 1) * PJ + n]);
}

__global__ __launch_bounds__(256) void pack_gen_k(const float* __restrict__ W,
                                                  unsigned* __restrict__ dst,
                                                  int kcCount, int ncols) {
    int g    = blockIdx.x * 256 + threadIdx.x;
    int j    = g & 7;
    int lane = (g >> 3) & 31;
    int frag = g >> 8;
    int kc   = frag % kcCount;
    int nt   = frag / kcCount;
    int n    = nt * 16 + (lane & 15);
    int k0   = kc * 32 + ((lane >> 4) << 4) + 2 * j;
    dst[g] = bf16pk(W[k0 * ncols + n], W[(k0 + 1) * ncols + n]);
}

// ---------------------------------------------------------------------------
// Fused kernel: one block = 32 batch rows. All LDS matrix buffers are stored
// pre-swizzled in WMMA A-fragment order [kc][h][lane][8 u32], so fragment
// fetch = 2x ds_load_b128 per fragment (lane-contiguous 32B).
// LDS map (bytes):
//   [0,      196608)  sA  : A fragments, 96 kc x 2 h x 256 u32
//   [196608, 229376)  sC1 : composed1 fragments, 16 kc x 2 h x 256 u32
//   after GEMM1 barrier, sA region reused:
//   [0,       32768)  sC2 : composed2 fragments
//   [32768,   36864)  sScore: 32 x 32 f32
//   [229376, 229632)  sRow: 32 row base pointers
// Total 229632 B (< 320 KB WGP budget).
// ---------------------------------------------------------------------------
__global__ __launch_bounds__(256) void relpred_fused(const float* __restrict__ enc,
                                                     const int* __restrict__ pidx,
                                                     const unsigned* __restrict__ wcat,
                                                     const unsigned* __restrict__ whid,
                                                     const unsigned* __restrict__ wsc,
                                                     float* __restrict__ out) {
    __shared__ __align__(16) char smem[229632];
    unsigned*     sA     = (unsigned*)(smem);
    unsigned*     sC1    = (unsigned*)(smem + 196608);
    unsigned*     sC2    = (unsigned*)(smem);
    float*        sScore = (float*)(smem + 32768);
    const float** sRow   = (const float**)(smem + 229376);

    const int tid  = threadIdx.x;
    const int ws   = __builtin_amdgcn_readfirstlane(tid >> 5);  // wave id, forced SGPR
    const int lane = tid & 31;
    const int hi   = lane >> 4;         // 0 / 1
    const int lo   = lane & 15;
    const int mt   = blockIdx.x;        // batch tile (32 rows)

    // Per-row gather base: rows idx-1, idx, idx+1 are contiguous (3072 floats)
    if (tid < MT) {
        int b  = mt * MT + tid;
        int ix = pidx[b];
        sRow[tid] = enc + ((size_t)b * SS + (ix - 1)) * DD;
    }
    __syncthreads();

    // Stage A: 32 x 3072 f32 -> bf16 fragments in LDS.
    // float4 covers pairs (2cl, 2cl+1) which land in adjacent slots -> one b64 store.
    for (int q = tid; q < MT * (K1 / 4); q += 256) {
        int r  = q / (K1 / 4);
        int c  = q % (K1 / 4);
        int kc = c >> 3, cl = c & 7;
        float4 f = ((const float4*)sRow[r])[c];
        uint2 u;
        u.x = bf16pk(f.x, f.y);
        u.y = bf16pk(f.z, f.w);
        int base = (kc * 2 + (r >> 4)) * 256;
        *(uint2*)&sA[base + aslot(r & 15, 2 * cl)] = u;
    }
    __syncthreads();

    // ---------------- GEMM1: composed1 = tanh(A @ Wcat), K=3072, N=512 -------
    {
        v8f acc[2][4];
#pragma unroll
        for (int h = 0; h < 2; ++h)
#pragma unroll
            for (int i = 0; i < 4; ++i)
#pragma unroll
                for (int r = 0; r < 8; ++r) acc[h][i][r] = 0.f;

        const uint4* pW = (const uint4*)wcat + (size_t)ws * (KC1 * 64) + (size_t)lane * 2;
        const uint4* pA = (const uint4*)sA + (size_t)lane * 2;
        for (int kc = 0; kc < KC1; ++kc) {
            __builtin_prefetch(pW + 64, 0, 0);   // next K-chunk, first N-tile
            Frag a0, a1;
            a0.q[0] = pA[0];  a0.q[1] = pA[1];   // h=0 fragment
            a1.q[0] = pA[64]; a1.q[1] = pA[65];  // h=1 fragment (+256 u32)
#pragma unroll
            for (int i = 0; i < 4; ++i) {
                Frag b;
                const uint4* p = pW + (size_t)i * (8 * KC1 * 64);
                b.q[0] = p[0];
                b.q[1] = p[1];
                acc[0][i] = __builtin_amdgcn_wmma_f32_16x16x32_bf16(
                    false, a0.v, false, b.v, (short)0, acc[0][i], false, false);
                acc[1][i] = __builtin_amdgcn_wmma_f32_16x16x32_bf16(
                    false, a1.v, false, b.v, (short)0, acc[1][i], false, false);
            }
            pW += 64;
            pA += 128;   // 512 u32 per K-chunk (2 fragments)
        }
        // tanh, pair adjacent-N lanes, store packed bf16 into fragment slots
#pragma unroll
        for (int h = 0; h < 2; ++h)
#pragma unroll
            for (int i = 0; i < 4; ++i) {
                int nt = ws + 8 * i;
#pragma unroll
                for (int r = 0; r < 8; ++r) {
                    float t = fast_tanh(acc[h][i][r]);
                    float p = __shfl_xor(t, 1);
                    if (!(lane & 1)) {
                        int ml = r + hi * 8;
                        int pg = nt * 8 + (lo >> 1);       // global pair index 0..255
                        sC1[((pg >> 4) * 2 + h) * 256 + aslot(ml, pg & 15)] = bf16pk(t, p);
                    }
                }
            }
    }
    __syncthreads();   // sA dead from here; sC2/sScore reuse its region

    // ---------------- GEMM2: composed2 = tanh(composed1 @ hidden), K=512 -----
    {
        v8f acc[2][4];
#pragma unroll
        for (int h = 0; h < 2; ++h)
#pragma unroll
            for (int i = 0; i < 4; ++i)
#pragma unroll
                for (int r = 0; r < 8; ++r) acc[h][i][r] = 0.f;

        const uint4* pW = (const uint4*)whid + (size_t)ws * (KC2 * 64) + (size_t)lane * 2;
        const uint4* pA = (const uint4*)sC1 + (size_t)lane * 2;
        for (int kc = 0; kc < KC2; ++kc) {
            Frag a0, a1;
            a0.q[0] = pA[0];  a0.q[1] = pA[1];
            a1.q[0] = pA[64]; a1.q[1] = pA[65];
#pragma unroll
            for (int i = 0; i < 4; ++i) {
                Frag b;
                const uint4* p = pW + (size_t)i * (8 * KC2 * 64);
                b.q[0] = p[0];
                b.q[1] = p[1];
                acc[0][i] = __builtin_amdgcn_wmma_f32_16x16x32_bf16(
                    false, a0.v, false, b.v, (short)0, acc[0][i], false, false);
                acc[1][i] = __builtin_amdgcn_wmma_f32_16x16x32_bf16(
                    false, a1.v, false, b.v, (short)0, acc[1][i], false, false);
            }
            pW += 64;
            pA += 128;
        }
#pragma unroll
        for (int h = 0; h < 2; ++h)
#pragma unroll
            for (int i = 0; i < 4; ++i) {
                int nt = ws + 8 * i;
#pragma unroll
                for (int r = 0; r < 8; ++r) {
                    float t = fast_tanh(acc[h][i][r]);
                    float p = __shfl_xor(t, 1);
                    if (!(lane & 1)) {
                        int ml = r + hi * 8;
                        int pg = nt * 8 + (lo >> 1);
                        sC2[((pg >> 4) * 2 + h) * 256 + aslot(ml, pg & 15)] = bf16pk(t, p);
                    }
                }
            }
    }
    __syncthreads();

    // ---------------- GEMM3: scores = composed2 @ scorer, K split over waves -
    for (int q = tid; q < MT * OUTN; q += 256) sScore[q] = 0.f;
    __syncthreads();
    {
        v8f acc[2][2];
#pragma unroll
        for (int h = 0; h < 2; ++h)
#pragma unroll
            for (int t = 0; t < 2; ++t)
#pragma unroll
                for (int r = 0; r < 8; ++r) acc[h][t][r] = 0.f;

        const uint4* pW = (const uint4*)wsc + (size_t)(ws * 2) * 64 + (size_t)lane * 2;
        const uint4* pA = (const uint4*)sC2 + (size_t)(ws * 2) * 128 + (size_t)lane * 2;
#pragma unroll
        for (int kk = 0; kk < 2; ++kk) {        // kc = ws*2 + kk
            Frag a0, a1;
            a0.q[0] = pA[kk * 128];      a0.q[1] = pA[kk * 128 + 1];
            a1.q[0] = pA[kk * 128 + 64]; a1.q[1] = pA[kk * 128 + 65];
#pragma unroll
            for (int t = 0; t < NTS; ++t) {
                Frag b;
                const uint4* p = pW + (size_t)kk * 64 + (size_t)t * (KC2 * 64);
                b.q[0] = p[0];
                b.q[1] = p[1];
                acc[0][t] = __builtin_amdgcn_wmma_f32_16x16x32_bf16(
                    false, a0.v, false, b.v, (short)0, acc[0][t], false, false);
                acc[1][t] = __builtin_amdgcn_wmma_f32_16x16x32_bf16(
                    false, a1.v, false, b.v, (short)0, acc[1][t], false, false);
            }
        }
#pragma unroll
        for (int h = 0; h < 2; ++h)
#pragma unroll
            for (int t = 0; t < NTS; ++t) {
                int n = t * 16 + lo;
#pragma unroll
                for (int r = 0; r < 8; ++r) {
                    int m = h * 16 + r + hi * 8;
                    atomicAdd(&sScore[m * OUTN + n], acc[h][t][r]);   // ds_add_f32
                }
            }
    }
    __syncthreads();

    // ---------------- softmax over 32 + writeout -----------------------------
    if (tid < MT) {
        float mx = -3.402823466e38f;
        for (int n = 0; n < OUTN; ++n) mx = fmaxf(mx, sScore[tid * OUTN + n]);
        float s = 0.f;
        for (int n = 0; n < OUTN; ++n) {
            float e = __expf(sScore[tid * OUTN + n] - mx);
            sScore[tid * OUTN + n] = e;
            s += e;
        }
        float inv = 1.f / s;
        float* o = out + ((size_t)mt * MT + tid) * OUTN;
        for (int n = 0; n < OUTN; ++n) o[n] = sScore[tid * OUTN + n] * inv;
    }
}

// ---------------------------------------------------------------------------
extern "C" void kernel_launch(void* const* d_in, const int* in_sizes, int n_in,
                              void* d_out, int out_size, void* d_ws, size_t ws_size,
                              hipStream_t stream) {
    (void)in_sizes; (void)n_in; (void)out_size; (void)ws_size;
    const float* enc  = (const float*)d_in[0];
    const int*   pidx = (const int*)d_in[1];
    const float* Wh   = (const float*)d_in[2];
    const float* Wp   = (const float*)d_in[3];
    const float* Wc   = (const float*)d_in[4];
    const float* Whd  = (const float*)d_in[5];
    const float* Wsc  = (const float*)d_in[6];

    unsigned* wcat = (unsigned*)d_ws;             // 3.0 MB
    unsigned* whid = wcat + WCAT_U;               // 0.5 MB
    unsigned* wsc  = whid + WHID_U;               // 32 KB

    pack_cat_k<<<(int)(WCAT_U / 256), 256, 0, stream>>>(Wh, Wp, Wc, wcat);
    pack_gen_k<<<(int)(WHID_U / 256), 256, 0, stream>>>(Whd, whid, KC2, PJ);
    pack_gen_k<<<(int)(WSC_U  / 256), 256, 0, stream>>>(Wsc, wsc, KC2, OUTN);

    relpred_fused<<<BB / MT, 256, 0, stream>>>(enc, pidx, wcat, whid, wsc, (float*)d_out);
}
